// EncoderLayer_66451734003853
// MI455X (gfx1250) — compile-verified, matrix operand order checked by reading, not run on hardware
//
#include <hip/hip_runtime.h>
#include <hip/hip_bf16.h>
#include <math.h>

// ---------------------------------------------------------------------------
// MI455X (gfx1250) encoder layer: all GEMMs on v_wmma_f32_16x16x32_bf16,
// global->LDS staging via async copies (ASYNCcnt) with LDS double buffering.
// wave32, 256-thread blocks (8 waves), 128x128x32 tiles.
// ---------------------------------------------------------------------------

typedef __attribute__((ext_vector_type(8)))  __bf16 bhalf8;
typedef __attribute__((ext_vector_type(16))) __bf16 bhalf16;
typedef __attribute__((ext_vector_type(8)))  float  v8f;

#define BM 128
#define BN 128
#define BK 32
#define LDSS (BK + 8)   // 40 bf16 per LDS row: 80B stride, keeps 16B chunks aligned

// Load one 16x16x32 bf16 fragment from an LDS tile stored row-major with
// stride LDSS.  Per the CDNA5 ISA 16-bit A/B layout: lane L holds row (L&15),
// K chunks [8*hi, 8*hi+8) and [16+8*hi, 16+8*hi+8) where hi = L>>4.
__device__ inline bhalf16 frag_ld(const __bf16* base) {
    int lane = threadIdx.x & 31;
    const __bf16* p = base + (lane & 15) * LDSS + ((lane >> 4) << 3);
    bhalf8 a = *(const bhalf8*)p;
    bhalf8 b = *(const bhalf8*)(p + 16);
    bhalf16 r;
#pragma unroll
    for (int i = 0; i < 8; ++i) { r[i] = a[i]; r[i + 8] = b[i]; }
    return r;
}

// Async global->LDS 16-byte copy (CDNA5, tracked by ASYNCcnt).
__device__ inline void async_b128(unsigned lds_off, const void* gptr) {
    asm volatile("global_load_async_to_lds_b128 %0, %1, off"
                 :: "v"(lds_off), "v"(gptr) : "memory");
}
__device__ inline void wait_async0() {
    asm volatile("s_wait_asynccnt 0x0" ::: "memory");
}

// MODE: 0 = QKV projection (+phi, scatter to head-major bf16)
//       1 = kv_sum split-K (transposed operand tiles, atomicAdd f32)
//       2 = num (store f32 interleaved c = e*8 + h)
//       3 = FFN1 (+bias, exact GELU, store bf16)
//       4 = FFN2 (+bias, store f32)
template <int MODE, bool TRANS>
__global__ __launch_bounds__(256) void gemm_k(
    const __bf16* __restrict__ A, const __bf16* __restrict__ Bt,
    int K, int lda, int ldb, long aZ, long bZ,
    float* __restrict__ out0, __bf16* __restrict__ out1,
    __bf16* __restrict__ out2, __bf16* __restrict__ out3,
    const float* __restrict__ bias, int Nn)
{
    __shared__ alignas(16) __bf16 lA[2][BM * LDSS];
    __shared__ alignas(16) __bf16 lB[2][BN * LDSS];

    const int tid  = threadIdx.x;
    const int lane = tid & 31;
    const int wave = tid >> 5;
    const int wm = (wave & 3) * 32;   // 4 waves down M
    const int wn = (wave >> 2) * 64;  // 2 waves across N

    A  += (long)blockIdx.z * aZ;
    Bt += (long)blockIdx.z * bZ;

    int m0, n0, kbeg;
    if (TRANS) { m0 = 0; n0 = 0; kbeg = blockIdx.x * K; }
    else       { m0 = blockIdx.x * BM; n0 = blockIdx.y * BN; kbeg = 0; }

    v8f acc[2][4] = {};

    // Issue the async fill of one K-tile into LDS buffer `buf`.
    auto issue_tile = [&](int k0, int buf) {
#pragma unroll
        for (int cid = tid; cid < 512; cid += 256) {
            int row = cid >> 2, off = (cid & 3) * 8;
            async_b128((unsigned)(size_t)&lA[buf][row * LDSS + off],
                       &A[(long)(m0 + row) * lda + k0 + off]);
            async_b128((unsigned)(size_t)&lB[buf][row * LDSS + off],
                       &Bt[(long)(n0 + row) * ldb + k0 + off]);
        }
    };

    // One K-step of WMMA math out of LDS buffer `buf`.
    auto compute = [&](int buf) {
        bhalf16 af[2], bfr[4];
#pragma unroll
        for (int mi = 0; mi < 2; ++mi)
            af[mi] = frag_ld(&lA[buf][(wm + mi * 16) * LDSS]);
#pragma unroll
        for (int ni = 0; ni < 4; ++ni)
            bfr[ni] = frag_ld(&lB[buf][(wn + ni * 16) * LDSS]);
#pragma unroll
        for (int mi = 0; mi < 2; ++mi)
#pragma unroll
            for (int ni = 0; ni < 4; ++ni)
                acc[mi][ni] = __builtin_amdgcn_wmma_f32_16x16x32_bf16(
                    false, af[mi], false, bfr[ni], (short)0, acc[mi][ni],
                    false, false);
    };

    if constexpr (!TRANS) {
        // Async double-buffered pipeline: one barrier per K-step; the
        // (t+1)-tile DMA overlaps the t-tile WMMA chain.
        const int nt = K / BK;
        issue_tile(kbeg, 0);
        for (int t = 0; t < nt; ++t) {
            const int buf = t & 1;
            wait_async0();        // own wave's async fills of buf complete
            __syncthreads();      // everyone's fills of buf visible
            if (t + 1 < nt) issue_tile(kbeg + (t + 1) * BK, buf ^ 1);
            compute(buf);
        }
    } else {
        // Both operands are N-major in memory: transpose through VGPRs.
        for (int kt = 0; kt < K; kt += BK) {
            const int k0 = kbeg + kt;
            __syncthreads();
#pragma unroll
            for (int cid = tid; cid < 512; cid += 256) {
                int kk = cid >> 4, dc = (cid & 15) * 8;
                bhalf8 va = *(const bhalf8*)&A[(long)(k0 + kk) * lda + dc];
                bhalf8 vb = *(const bhalf8*)&Bt[(long)(k0 + kk) * ldb + dc];
#pragma unroll
                for (int j = 0; j < 8; ++j) {
                    lA[0][(dc + j) * LDSS + kk] = va[j];
                    lB[0][(dc + j) * LDSS + kk] = vb[j];
                }
            }
            if (kt + BK < K) { // prefetch next K-tile (global_prefetch_b8)
                __builtin_prefetch(&A[(long)(k0 + BK + (tid >> 4)) * lda], 0, 0);
                __builtin_prefetch(&Bt[(long)(k0 + BK + (tid >> 4)) * ldb], 0, 0);
            }
            __syncthreads();
            compute(0);
        }
    }

    // Epilogue: D layout — VGPR r holds M = r + 8*hi, N = lane & 15.
    const int hi = lane >> 4, cl = lane & 15;
#pragma unroll
    for (int mi = 0; mi < 2; ++mi)
#pragma unroll
        for (int ni = 0; ni < 4; ++ni)
#pragma unroll
            for (int r = 0; r < 8; ++r) {
                int row = m0 + wm + mi * 16 + r + 8 * hi;
                int col = n0 + wn + ni * 16 + cl;
                float v = acc[mi][ni][r];
                if constexpr (MODE == 0) {
                    int seg = col >> 10;       // 0=q 1=k 2=v
                    int c = col & 1023;
                    int h = c & 7, i2 = c >> 3;
                    long o = ((long)h * Nn + row) * 128 + i2;
                    if (seg == 0) {            // phi(q) = elu(q)+1
                        out1[o] = (__bf16)(v > 0.f ? v + 1.f : __expf(v));
                    } else if (seg == 1) {     // phi(k)
                        out2[o] = (__bf16)(v > 0.f ? v + 1.f : __expf(v));
                    } else {                   // v
                        out3[o] = (__bf16)v;
                    }
                } else if constexpr (MODE == 1) {
                    atomicAdd(&out0[((long)blockIdx.z << 14) + row * 128 + col], v);
                } else if constexpr (MODE == 2) {
                    out0[(long)row * 1024 + col * 8 + blockIdx.z] = v;  // c = e*8+h
                } else if constexpr (MODE == 3) {
                    v += bias[col];
                    float g = 0.5f * v * (1.f + erff(v * 0.70710678118f));
                    out1[(long)row * 1024 + col] = (__bf16)g;
                } else {  // MODE 4
                    out0[(long)row * 1024 + col] = v + bias[col];
                }
            }
}

// --------------------------- small helper kernels ---------------------------

__global__ void cvt_f32_bf16(const float* __restrict__ s, __bf16* __restrict__ d) {
    long i = ((long)blockIdx.x * blockDim.x + threadIdx.x) * 4;
    float4 v = *(const float4*)(s + i);
    d[i + 0] = (__bf16)v.x; d[i + 1] = (__bf16)v.y;
    d[i + 2] = (__bf16)v.z; d[i + 3] = (__bf16)v.w;
}

// dst[m*K + k] = (bf16) src[k*M + m]   (transpose-convert weights to K-major)
__global__ void tconv(const float* __restrict__ src, __bf16* __restrict__ dst,
                      int K, int M) {
    long idx = (long)blockIdx.x * blockDim.x + threadIdx.x;
    long m = idx / K, k = idx - m * K;
    if (m < M) dst[idx] = (__bf16)src[k * (long)M + m];
}

__global__ void zerof(float* p, long n) {
    long i = (long)blockIdx.x * blockDim.x + threadIdx.x;
    if (i < n) p[i] = 0.f;
}

// z[h][i] = sum_n phiK[h][n][i]
__global__ void zsum_k(const __bf16* __restrict__ phiK, float* __restrict__ z, int Nn) {
    int h = blockIdx.x, split = blockIdx.y;
    int i = threadIdx.x & 127, sub = threadIdx.x >> 7;
    const __bf16* p = phiK + (long)h * Nn * 128 + i;
    int chunk = Nn / 32, n0 = split * chunk;
    float s = 0.f;
    for (int n = n0 + sub; n < n0 + chunk; n += 2) s += (float)p[(long)n * 128];
    atomicAdd(&z[h * 128 + i], s);
}

// kvbt[h][e][d] = (bf16) kvsum[h][d][e]   (K-major for the num GEMM)
__global__ void cvt_kvsum_k(const float* __restrict__ kvsum, __bf16* __restrict__ kvbt) {
    int idx = blockIdx.x * 256 + threadIdx.x;
    int h = idx >> 14, rem = idx & 16383, d = rem >> 7, e = rem & 127;
    kvbt[(h << 14) + e * 128 + d] = (__bf16)kvsum[idx];
}

__device__ inline float block_reduce(float v, float* red) {
    int t = threadIdx.x;
    red[t] = v; __syncthreads();
#pragma unroll
    for (int s = 128; s > 0; s >>= 1) {
        if (t < s) red[t] += red[t + s];
        __syncthreads();
    }
    float r = red[0]; __syncthreads();
    return r;
}

// t = num/(den+eps); u = t + x; y = LN1(u); also y in bf16 for FFN
__global__ __launch_bounds__(256) void addnorm1_k(
    const float* __restrict__ num, const float* __restrict__ x,
    const __bf16* __restrict__ phiQ, const float* __restrict__ z,
    const float* __restrict__ sc, const float* __restrict__ bi,
    float* __restrict__ y, __bf16* __restrict__ yb, int Nn)
{
    __shared__ float den_s[8];
    __shared__ float red[256];
    const int n = blockIdx.x, t = threadIdx.x;
    if (t < 8) den_s[t] = 0.f;
    __syncthreads();

    float4 xv = *(const float4*)&x[(long)n * 1024 + t * 4];
    float4 nv = *(const float4*)&num[(long)n * 1024 + t * 4];
#pragma unroll
    for (int j = 0; j < 4; ++j) {
        int c = t * 4 + j, h = c & 7, i2 = c >> 3;
        float pq = (float)phiQ[((long)h * Nn + n) * 128 + i2];
        atomicAdd(&den_s[h], pq * z[h * 128 + i2]);
    }
    __syncthreads();

    float u[4];
    const float* nvp = &nv.x; const float* xvp = &xv.x;
#pragma unroll
    for (int j = 0; j < 4; ++j) {
        int c = t * 4 + j, h = c & 7;
        u[j] = nvp[j] / (den_s[h] + 1e-6f) + xvp[j];
    }
    float mu = block_reduce(u[0] + u[1] + u[2] + u[3], red) * (1.f / 1024.f);
    float vs = 0.f;
#pragma unroll
    for (int j = 0; j < 4; ++j) { float dd = u[j] - mu; vs += dd * dd; }
    float var = block_reduce(vs, red) * (1.f / 1024.f);
    float rstd = rsqrtf(var + 1e-5f);
#pragma unroll
    for (int j = 0; j < 4; ++j) {
        int c = t * 4 + j;
        float o = (u[j] - mu) * rstd * sc[c] + bi[c];
        y[(long)n * 1024 + c]  = o;
        yb[(long)n * 1024 + c] = (__bf16)o;
    }
}

// out = LN2(f + y)
__global__ __launch_bounds__(256) void addnorm2_k(
    const float* __restrict__ f, const float* __restrict__ y,
    const float* __restrict__ sc, const float* __restrict__ bi,
    float* __restrict__ out)
{
    __shared__ float red[256];
    const int n = blockIdx.x, t = threadIdx.x;
    float4 fv = *(const float4*)&f[(long)n * 1024 + t * 4];
    float4 yv = *(const float4*)&y[(long)n * 1024 + t * 4];
    float u[4] = { fv.x + yv.x, fv.y + yv.y, fv.z + yv.z, fv.w + yv.w };
    float mu = block_reduce(u[0] + u[1] + u[2] + u[3], red) * (1.f / 1024.f);
    float vs = 0.f;
#pragma unroll
    for (int j = 0; j < 4; ++j) { float dd = u[j] - mu; vs += dd * dd; }
    float var = block_reduce(vs, red) * (1.f / 1024.f);
    float rstd = rsqrtf(var + 1e-5f);
#pragma unroll
    for (int j = 0; j < 4; ++j) {
        int c = t * 4 + j;
        out[(long)n * 1024 + c] = (u[j] - mu) * rstd * sc[c] + bi[c];
    }
}

// ---------------------------------------------------------------------------

extern "C" void kernel_launch(void* const* d_in, const int* in_sizes, int n_in,
                              void* d_out, int out_size, void* d_ws, size_t ws_size,
                              hipStream_t stream) {
    const float* x    = (const float*)d_in[0];
    const float* Wq   = (const float*)d_in[1];
    const float* Wkv  = (const float*)d_in[2];
    const float* ln1s = (const float*)d_in[3];
    const float* ln1b = (const float*)d_in[4];
    const float* W1   = (const float*)d_in[5];
    const float* b1   = (const float*)d_in[6];
    const float* W2   = (const float*)d_in[7];
    const float* b2   = (const float*)d_in[8];
    const float* ln2s = (const float*)d_in[9];
    const float* ln2b = (const float*)d_in[10];
    float* out = (float*)d_out;

    const int Nn = 16384, C = 1024, H = 8, D = 128;

    char* w = (char*)d_ws;
    auto alloc = [&](size_t bytes) -> char* {
        char* p = w; w += (bytes + 255) & ~(size_t)255; return p;
    };
    __bf16* xb    = (__bf16*)alloc((size_t)Nn * C * 2);
    __bf16* Wqkvt = (__bf16*)alloc((size_t)3 * C * C * 2);
    __bf16* W1t   = (__bf16*)alloc((size_t)C * C * 2);
    __bf16* W2t   = (__bf16*)alloc((size_t)C * C * 2);
    __bf16* phiQ  = (__bf16*)alloc((size_t)Nn * C * 2);
    __bf16* phiK  = (__bf16*)alloc((size_t)Nn * C * 2);
    __bf16* vb    = (__bf16*)alloc((size_t)Nn * C * 2);
    float*  z     = (float*)alloc((size_t)(H * D + H * D * D) * 4);
    float*  kvsum = z + H * D;
    __bf16* kvbt  = (__bf16*)alloc((size_t)H * D * D * 2);
    float*  num   = (float*)alloc((size_t)Nn * C * 4);   // reused as FFN2 output f
    float*  y     = (float*)alloc((size_t)Nn * C * 4);
    __bf16* yb    = (__bf16*)alloc((size_t)Nn * C * 2);
    __bf16* hb    = (__bf16*)alloc((size_t)Nn * C * 2);

    // Precision conversion + weight transposition (K-major operands for WMMA)
    cvt_f32_bf16<<<Nn * C / 1024, 256, 0, stream>>>(x, xb);
    tconv<<<C * C / 256, 256, 0, stream>>>(Wq, Wqkvt, C, C);
    tconv<<<C * 2 * C / 256, 256, 0, stream>>>(Wkv, Wqkvt + (size_t)C * C, C, 2 * C);
    tconv<<<C * C / 256, 256, 0, stream>>>(W1, W1t, C, C);
    tconv<<<C * C / 256, 256, 0, stream>>>(W2, W2t, C, C);
    zerof<<<(H * D + H * D * D + 255) / 256, 256, 0, stream>>>(z, H * D + H * D * D);

    // QKV projection + phi, scattered head-major: [H][N][128] bf16
    gemm_k<0, false><<<dim3(Nn / BM, 3 * C / BN, 1), 256, 0, stream>>>(
        xb, Wqkvt, C, C, C, 0, 0, nullptr, phiQ, phiK, vb, nullptr, Nn);

    // z[h][d] = sum_n phiK
    zsum_k<<<dim3(H, 32), 256, 0, stream>>>(phiK, z, Nn);

    // kv_sum[h][d][e] = phiK_h^T @ v_h  (split-K over N, atomic f32 accumulate)
    gemm_k<1, true><<<dim3(32, 1, H), 256, 0, stream>>>(
        phiK, vb, Nn / 32, D, D, (long)Nn * D, (long)Nn * D,
        kvsum, nullptr, nullptr, nullptr, nullptr, Nn);

    cvt_kvsum_k<<<H * D * D / 256, 256, 0, stream>>>(kvsum, kvbt);

    // num[n][e*8+h] = phiQ_h @ kv_sum_h
    gemm_k<2, false><<<dim3(Nn / BM, 1, H), 256, 0, stream>>>(
        phiQ, kvbt, D, D, D, (long)Nn * D, (long)D * D,
        num, nullptr, nullptr, nullptr, nullptr, Nn);

    // t = num/(den+eps); y = LN1(t + x)
    addnorm1_k<<<Nn, 256, 0, stream>>>(num, x, phiQ, z, ln1s, ln1b, y, yb, Nn);

    // FFN
    gemm_k<3, false><<<dim3(Nn / BM, C / BN, 1), 256, 0, stream>>>(
        yb, W1t, C, C, C, 0, 0, nullptr, hb, nullptr, nullptr, b1, Nn);
    gemm_k<4, false><<<dim3(Nn / BM, C / BN, 1), 256, 0, stream>>>(
        hb, W2t, C, C, C, 0, 0, num, nullptr, nullptr, nullptr, b2, Nn);

    // out = LN2(f + y)
    addnorm2_k<<<Nn, 256, 0, stream>>>(num, y, ln2s, ln2b, out);
}
